// CA_82300163326040
// MI455X (gfx1250) — matrix operand order repaired and checked by reading, not compiled, not run on hardware
//

#include <hip/hip_runtime.h>
#include <hip/hip_bf16.h>
#include <math.h>

#define Bsz 8
#define Nn  2048
#define Dd  512

#define LDS_STRIDE 40   // u16 elems per LDS tile row: 32 data + 8 pad = 80B (16B aligned, conflict-free)

typedef __attribute__((ext_vector_type(16))) __bf16 bf16x16;
typedef __attribute__((ext_vector_type(8)))  float  f32x8;
typedef __attribute__((ext_vector_type(4)))  unsigned int u32x4;
typedef unsigned short u16;
typedef unsigned int   u32;

__device__ __forceinline__ u16 f32_to_bf16(float f) {
  union { float f; u32 u; } v; v.f = f;
  u32 u = v.u;
  u32 r = u + 0x7FFFu + ((u >> 16) & 1u);      // round-to-nearest-even
  if ((u & 0x7F800000u) == 0x7F800000u) r = u; // inf/nan: truncate
  return (u16)(r >> 16);
}

__device__ __forceinline__ float bf16_to_f32(u16 h) {
  union { u32 u; float f; } v; v.u = ((u32)h) << 16;
  return v.f;
}

__device__ __forceinline__ f32x8 wmma_bf16(bf16x16 a, bf16x16 b, f32x8 c) {
  return __builtin_amdgcn_wmma_f32_16x16x32_bf16(false, a, false, b, (short)0, c,
                                                 false, false);
}

// LDS byte address of a __shared__-derived pointer (flat shared addr low 32 bits = LDS offset)
__device__ __forceinline__ u32 lds_addr(const void* p) { return (u32)(uintptr_t)p; }

// Async copy of a ROWSx32 bf16 tile (row-major source, stride ldg elems) into LDS
// (row stride LDS_STRIDE elems). One 16B chunk per lane per issue: ROWS*4 chunks total.
template <int ROWS>
__device__ __forceinline__ void async_copy_tile(u32 lds_off, const u16* __restrict__ gbase,
                                                long row0, int ldg, int k0, int tid) {
#pragma unroll
  for (int c = tid; c < ROWS * 4; c += 256) {
    int row = c >> 2, seg = c & 3;
    const u16* g = gbase + (size_t)(row0 + row) * (size_t)ldg + (size_t)(k0 + seg * 8);
    u32 l = lds_off + (u32)(row * (LDS_STRIDE * 2) + seg * 16);
    asm volatile("global_load_async_to_lds_b128 %0, %1, off" :: "v"(l), "v"(g) : "memory");
  }
}

__device__ __forceinline__ void wait_async_all() {
  asm volatile("s_wait_asynccnt 0" ::: "memory");
}

// A fragment (16x32 bf16) from LDS tile (rows = M, 32 K's, stride LDS_STRIDE).
// lanes 0-15: M=lane, K 0..15 ; lanes 16-31: M=lane-16, K 16..31.
__device__ __forceinline__ bf16x16 lds_frag_a(const u16* tile, int row0, int lane) {
  int half = lane >> 4, r = lane & 15;
  const u16* p = tile + (row0 + r) * LDS_STRIDE;
  union { bf16x16 v; u32x4 q[2]; } f;
  f.q[0] = *(const u32x4*)(p + half * 16);
  f.q[1] = *(const u32x4*)(p + half * 16 + 8);
  return f.v;
}

// B fragment (32x16): B[k][col] = tile[col][k]. lanes 0-15: K 0..15; lanes 16-31: K 16..31.
__device__ __forceinline__ bf16x16 lds_frag_b(const u16* tile, int col0, int lane) {
  int half = lane >> 4, r = lane & 15;
  const u16* p = tile + (col0 + r) * LDS_STRIDE + half * 16;
  union { bf16x16 v; u32x4 q[2]; } f;
  f.q[0] = *(const u32x4*)(p);
  f.q[1] = *(const u32x4*)(p + 8);
  return f.v;
}

// ---------------- conversion f32 -> bf16 ----------------
__global__ void k_cvt_bf16(const float* __restrict__ src, u16* __restrict__ dst, int n) {
  int i = blockIdx.x * blockDim.x + threadIdx.x;
  if (i < n) dst[i] = f32_to_bf16(src[i]);
}

// ---------------- projection: Out = In @ W^T + bias ----------------
// In bf16 [B*N, D]; W bf16 [D(e), D(d)]; bias f32 [D].
// Macro tile 256x128, 8 waves of 32x128 each; LDS double-buffered, async-filled.
__global__ __launch_bounds__(256) void k_proj(const u16* __restrict__ A,
                                              const u16* __restrict__ W,
                                              const float* __restrict__ bias,
                                              u16* __restrict__ out, int transposed) {
  __shared__ __align__(16) u16 As[2][256 * LDS_STRIDE];
  __shared__ __align__(16) u16 Bs[2][128 * LDS_STRIDE];
  int tid = threadIdx.x, wid = tid >> 5, lane = tid & 31;
  long row0 = (long)blockIdx.y * 256;   // flattened B*N rows
  int col0 = blockIdx.x * 128;
  int wr = wid;                         // wave: rows wr*32..

  const f32x8 z = {0.f, 0.f, 0.f, 0.f, 0.f, 0.f, 0.f, 0.f};
  f32x8 acc[2][8] = {{z, z, z, z, z, z, z, z}, {z, z, z, z, z, z, z, z}};

  async_copy_tile<256>(lds_addr(&As[0][0]), A, row0, Dd, 0, tid);
  async_copy_tile<128>(lds_addr(&Bs[0][0]), W, col0, Dd, 0, tid);

  const int KS = Dd / 32;  // 16
  for (int ks = 0; ks < KS; ++ks) {
    wait_async_all();
    __syncthreads();
    int buf = ks & 1;
    if (ks + 1 < KS) {
      async_copy_tile<256>(lds_addr(&As[buf ^ 1][0]), A, row0, Dd, (ks + 1) * 32, tid);
      async_copy_tile<128>(lds_addr(&Bs[buf ^ 1][0]), W, col0, Dd, (ks + 1) * 32, tid);
    }
    bf16x16 a0 = lds_frag_a(&As[buf][0], wr * 32, lane);
    bf16x16 a1 = lds_frag_a(&As[buf][0], wr * 32 + 16, lane);
#pragma unroll
    for (int t = 0; t < 8; ++t) {
      bf16x16 bf = lds_frag_b(&Bs[buf][0], t * 16, lane);
      acc[0][t] = wmma_bf16(a0, bf, acc[0][t]);
      acc[1][t] = wmma_bf16(a1, bf, acc[1][t]);
    }
  }

  int half = lane >> 4, r = lane & 15;
#pragma unroll
  for (int t = 0; t < 8; ++t) {
    int colL = col0 + t * 16 + r;
    float bv = bias[colL];
#pragma unroll
    for (int a = 0; a < 2; ++a) {
#pragma unroll
      for (int q = 0; q < 8; ++q) {
        long row = row0 + wr * 32 + a * 16 + q + 8 * half;
        float v = acc[a][t][q] + bv;
        if (!transposed) {
          out[(size_t)row * Dd + colL] = f32_to_bf16(v);
        } else {
          int b = (int)(row / Nn), n = (int)(row % Nn);
          out[((size_t)b * Dd + colL) * Nn + n] = f32_to_bf16(v);
        }
      }
    }
  }
}

// ---------------- scores: S[b,n,m] = scale * sum_d Q[b,n,d] K[b,m,d] ----------------
// Macro tile 256x128, 8 waves of 32x128.
__global__ __launch_bounds__(256) void k_score(const u16* __restrict__ Q,
                                               const u16* __restrict__ K,
                                               u16* __restrict__ S, float scale) {
  __shared__ __align__(16) u16 As[2][256 * LDS_STRIDE];
  __shared__ __align__(16) u16 Bs[2][128 * LDS_STRIDE];
  int tid = threadIdx.x, wid = tid >> 5, lane = tid & 31;
  long row0 = (long)blockIdx.y * 256;         // flattened Q rows (b*N + n)
  int col0 = blockIdx.x * 128;                // m
  int b = (int)(row0 / Nn);
  long brow0 = (long)b * Nn + col0;           // K rows for this block
  int wr = wid;

  const f32x8 z = {0.f, 0.f, 0.f, 0.f, 0.f, 0.f, 0.f, 0.f};
  f32x8 acc[2][8] = {{z, z, z, z, z, z, z, z}, {z, z, z, z, z, z, z, z}};

  async_copy_tile<256>(lds_addr(&As[0][0]), Q, row0, Dd, 0, tid);
  async_copy_tile<128>(lds_addr(&Bs[0][0]), K, brow0, Dd, 0, tid);

  const int KS = Dd / 32;
  for (int ks = 0; ks < KS; ++ks) {
    wait_async_all();
    __syncthreads();
    int buf = ks & 1;
    if (ks + 1 < KS) {
      async_copy_tile<256>(lds_addr(&As[buf ^ 1][0]), Q, row0, Dd, (ks + 1) * 32, tid);
      async_copy_tile<128>(lds_addr(&Bs[buf ^ 1][0]), K, brow0, Dd, (ks + 1) * 32, tid);
    }
    bf16x16 a0 = lds_frag_a(&As[buf][0], wr * 32, lane);
    bf16x16 a1 = lds_frag_a(&As[buf][0], wr * 32 + 16, lane);
#pragma unroll
    for (int t = 0; t < 8; ++t) {
      bf16x16 bf = lds_frag_b(&Bs[buf][0], t * 16, lane);
      acc[0][t] = wmma_bf16(a0, bf, acc[0][t]);
      acc[1][t] = wmma_bf16(a1, bf, acc[1][t]);
    }
  }

  int half = lane >> 4, r = lane & 15;
  size_t sb = (size_t)b * Nn * Nn;
#pragma unroll
  for (int t = 0; t < 8; ++t) {
    int m = col0 + t * 16 + r;
#pragma unroll
    for (int a = 0; a < 2; ++a) {
#pragma unroll
      for (int q = 0; q < 8; ++q) {
        long n = (row0 % Nn) + wr * 32 + a * 16 + q + 8 * half;
        S[sb + (size_t)n * Nn + m] = f32_to_bf16(acc[a][t][q] * scale);
      }
    }
  }
}

// ---------------- softmax over batch axis (size 8), in place ----------------
__global__ void k_softmax_b(u16* __restrict__ S) {
  size_t i = (size_t)blockIdx.x * blockDim.x + threadIdx.x;
  const size_t NN = (size_t)Nn * Nn;
  if (i >= NN) return;
  float v[Bsz];
  float mx = -INFINITY;
#pragma unroll
  for (int b = 0; b < Bsz; ++b) {
    v[b] = bf16_to_f32(S[(size_t)b * NN + i]);
    mx = fmaxf(mx, v[b]);
  }
  float sum = 0.f;
#pragma unroll
  for (int b = 0; b < Bsz; ++b) {
    v[b] = __expf(v[b] - mx);
    sum += v[b];
  }
  float inv = 1.f / sum;
#pragma unroll
  for (int b = 0; b < Bsz; ++b) S[(size_t)b * NN + i] = f32_to_bf16(v[b] * inv);
}

// ---------------- out = X + Y + A1@V2 + A2@V1 (shared accumulator) ----------------
// A1,A2 bf16 [B,N,N]; V2t,V1t bf16 [B,D,N]; X,Y,out f32 [B,N,D]
// Macro tile 128x128, 8 waves of 32x64; 4 double-buffered LDS tiles (80KB).
__global__ __launch_bounds__(256) void k_out(const u16* __restrict__ A1,
                                             const u16* __restrict__ V2t,
                                             const u16* __restrict__ A2,
                                             const u16* __restrict__ V1t,
                                             const float* __restrict__ X,
                                             const float* __restrict__ Yv,
                                             float* __restrict__ out) {
  __shared__ __align__(16) u16 A1s[2][128 * LDS_STRIDE];
  __shared__ __align__(16) u16 A2s[2][128 * LDS_STRIDE];
  __shared__ __align__(16) u16 B1s[2][128 * LDS_STRIDE];
  __shared__ __align__(16) u16 B2s[2][128 * LDS_STRIDE];
  int tid = threadIdx.x, wid = tid >> 5, lane = tid & 31;
  long row0 = (long)blockIdx.y * 128;  // flattened attention rows (b*N + n)
  int col0 = blockIdx.x * 128;         // d
  int b = (int)(row0 / Nn);
  long vrow0 = (long)b * Dd + col0;    // Vt rows for this block
  int wr = wid & 3, wc = wid >> 2;     // wave: rows wr*32.., cols wc*64..

  const f32x8 z = {0.f, 0.f, 0.f, 0.f, 0.f, 0.f, 0.f, 0.f};
  f32x8 acc[2][4] = {{z, z, z, z}, {z, z, z, z}};

  async_copy_tile<128>(lds_addr(&A1s[0][0]), A1, row0, Nn, 0, tid);
  async_copy_tile<128>(lds_addr(&A2s[0][0]), A2, row0, Nn, 0, tid);
  async_copy_tile<128>(lds_addr(&B1s[0][0]), V2t, vrow0, Nn, 0, tid);
  async_copy_tile<128>(lds_addr(&B2s[0][0]), V1t, vrow0, Nn, 0, tid);

  const int KS = Nn / 32;  // 64
  for (int ks = 0; ks < KS; ++ks) {
    wait_async_all();
    __syncthreads();
    int buf = ks & 1;
    if (ks + 1 < KS) {
      int kk = (ks + 1) * 32;
      async_copy_tile<128>(lds_addr(&A1s[buf ^ 1][0]), A1, row0, Nn, kk, tid);
      async_copy_tile<128>(lds_addr(&A2s[buf ^ 1][0]), A2, row0, Nn, kk, tid);
      async_copy_tile<128>(lds_addr(&B1s[buf ^ 1][0]), V2t, vrow0, Nn, kk, tid);
      async_copy_tile<128>(lds_addr(&B2s[buf ^ 1][0]), V1t, vrow0, Nn, kk, tid);
    }
    bf16x16 a1lo = lds_frag_a(&A1s[buf][0], wr * 32, lane);
    bf16x16 a1hi = lds_frag_a(&A1s[buf][0], wr * 32 + 16, lane);
    bf16x16 a2lo = lds_frag_a(&A2s[buf][0], wr * 32, lane);
    bf16x16 a2hi = lds_frag_a(&A2s[buf][0], wr * 32 + 16, lane);
#pragma unroll
    for (int t = 0; t < 4; ++t) {
      bf16x16 b1 = lds_frag_b(&B1s[buf][0], wc * 64 + t * 16, lane);
      acc[0][t] = wmma_bf16(a1lo, b1, acc[0][t]);
      acc[1][t] = wmma_bf16(a1hi, b1, acc[1][t]);
      bf16x16 b2 = lds_frag_b(&B2s[buf][0], wc * 64 + t * 16, lane);
      acc[0][t] = wmma_bf16(a2lo, b2, acc[0][t]);
      acc[1][t] = wmma_bf16(a2hi, b2, acc[1][t]);
    }
  }

  int half = lane >> 4, r = lane & 15;
#pragma unroll
  for (int t = 0; t < 4; ++t) {
    int col = col0 + wc * 64 + t * 16 + r;
#pragma unroll
    for (int a = 0; a < 2; ++a) {
#pragma unroll
      for (int q = 0; q < 8; ++q) {
        long row = row0 + wr * 32 + a * 16 + q + 8 * half;   // b*N + n
        size_t idx = (size_t)row * Dd + col;
        out[idx] = acc[a][t][q] + X[idx] + Yv[idx];
      }
    }
  }
}

extern "C" void kernel_launch(void* const* d_in, const int* in_sizes, int n_in,
                              void* d_out, int out_size, void* d_ws, size_t ws_size,
                              hipStream_t stream) {
  (void)in_sizes; (void)n_in; (void)out_size; (void)ws_size;

  const float* X    = (const float*)d_in[0];
  const float* Y    = (const float*)d_in[1];
  const float* W_xq = (const float*)d_in[2];  const float* b_xq = (const float*)d_in[3];
  const float* W_xk = (const float*)d_in[4];  const float* b_xk = (const float*)d_in[5];
  const float* W_xv = (const float*)d_in[6];  const float* b_xv = (const float*)d_in[7];
  const float* W_yq = (const float*)d_in[8];  const float* b_yq = (const float*)d_in[9];
  const float* W_yk = (const float*)d_in[10]; const float* b_yk = (const float*)d_in[11];
  const float* W_yv = (const float*)d_in[12]; const float* b_yv = (const float*)d_in[13];
  float* out = (float*)d_out;

  const size_t PB  = (size_t)Bsz * Nn * Dd;  // 8,388,608 elems
  const size_t WBe = (size_t)Dd * Dd;        // 262,144 elems
  const size_t NNB = (size_t)Bsz * Nn * Nn;  // 33,554,432 elems

  u16* ws   = (u16*)d_ws;
  u16* Xb   = ws;
  u16* Yb   = Xb + PB;
  u16* Wq1b = Yb + PB;
  u16* Wk1b = Wq1b + WBe;
  u16* Wv1b = Wk1b + WBe;
  u16* Wq2b = Wv1b + WBe;
  u16* Wk2b = Wq2b + WBe;
  u16* Wv2b = Wk2b + WBe;
  u16* Q1   = Wv2b + WBe;
  u16* K1   = Q1 + PB;
  u16* V1t  = K1 + PB;
  u16* Q2   = V1t + PB;
  u16* K2   = Q2 + PB;
  u16* V2t  = K2 + PB;
  u16* S1   = V2t + PB;   // bf16 logits -> attention weights, in place
  u16* S2   = S1 + NNB;

  // 1) f32 -> bf16 conversions
  k_cvt_bf16<<<(int)((PB + 255) / 256), 256, 0, stream>>>(X, Xb, (int)PB);
  k_cvt_bf16<<<(int)((PB + 255) / 256), 256, 0, stream>>>(Y, Yb, (int)PB);
  k_cvt_bf16<<<(int)((WBe + 255) / 256), 256, 0, stream>>>(W_xq, Wq1b, (int)WBe);
  k_cvt_bf16<<<(int)((WBe + 255) / 256), 256, 0, stream>>>(W_xk, Wk1b, (int)WBe);
  k_cvt_bf16<<<(int)((WBe + 255) / 256), 256, 0, stream>>>(W_xv, Wv1b, (int)WBe);
  k_cvt_bf16<<<(int)((WBe + 255) / 256), 256, 0, stream>>>(W_yq, Wq2b, (int)WBe);
  k_cvt_bf16<<<(int)((WBe + 255) / 256), 256, 0, stream>>>(W_yk, Wk2b, (int)WBe);
  k_cvt_bf16<<<(int)((WBe + 255) / 256), 256, 0, stream>>>(W_yv, Wv2b, (int)WBe);

  // 2) projections: grid (D/128, B*N/256) = (4, 64)
  dim3 pgrid(Dd / 128, (Bsz * Nn) / 256);
  k_proj<<<pgrid, 256, 0, stream>>>(Xb, Wq1b, b_xq, Q1, 0);
  k_proj<<<pgrid, 256, 0, stream>>>(Xb, Wk1b, b_xk, K1, 0);
  k_proj<<<pgrid, 256, 0, stream>>>(Xb, Wv1b, b_xv, V1t, 1);
  k_proj<<<pgrid, 256, 0, stream>>>(Yb, Wq2b, b_yq, Q2, 0);
  k_proj<<<pgrid, 256, 0, stream>>>(Yb, Wk2b, b_yk, K2, 0);
  k_proj<<<pgrid, 256, 0, stream>>>(Yb, Wv2b, b_yv, V2t, 1);

  // 3) scores: grid (N/128, B*N/256) = (16, 64)
  const float scale = 0.04419417382415922f;  // 1/sqrt(512)
  dim3 sgrid(Nn / 128, (Bsz * Nn) / 256);
  k_score<<<sgrid, 256, 0, stream>>>(Q1, K2, S1, scale);
  k_score<<<sgrid, 256, 0, stream>>>(Q2, K1, S2, scale);

  // 4) batch-axis softmax, in place: N*N threads
  const int smgrid = (int)(((size_t)Nn * Nn + 255) / 256);  // 16384
  k_softmax_b<<<smgrid, 256, 0, stream>>>(S1);
  k_softmax_b<<<smgrid, 256, 0, stream>>>(S2);

  // 5) fused A1@V2 + A2@V1 + X + Y: grid (D/128, B*N/128) = (4, 128)
  dim3 ogrid(Dd / 128, (Bsz * Nn) / 128);
  k_out<<<ogrid, 256, 0, stream>>>(S1, V2t, S2, V1t, X, Y, out);
}
